// GEDModelHomology_77575699300932
// MI455X (gfx1250) — matrix operand order repaired and checked by reading, not compiled, not run on hardware
//
#include <hip/hip_runtime.h>
#include <hip/hip_bf16.h>
#include <math.h>

#define F_DIM 2048
#define E_DIM 256
#define L1 33
#define CELLS (L1 * L1)       // 1089
#define P_PAIRS 256
#define INF_COST 10000.0f
#define BIG 1e30f

#define KBLK 64               // K-block staged per async round
#define BROW 72               // LDS row stride in halves (144B -> bank-spread)

typedef __attribute__((ext_vector_type(16))) _Float16 v16h;
typedef __attribute__((ext_vector_type(8)))  _Float16 v8h;
typedef __attribute__((ext_vector_type(8)))  float    v8f;

// ---------------------------------------------------------------------------
// fp32 -> fp16 weight conversion
// ---------------------------------------------------------------------------
__global__ void f32_to_f16_kernel(const float* __restrict__ src,
                                  _Float16* __restrict__ dst, int n) {
  int i = blockIdx.x * blockDim.x + threadIdx.x;
  if (i < n) dst[i] = (_Float16)src[i];
}

// ---------------------------------------------------------------------------
// Build padded fp16 activation matrix: [x_s ; x_t ; virtual_embedding ; 0-pad]
// ---------------------------------------------------------------------------
__global__ void build_x_kernel(const float* __restrict__ xs,
                               const float* __restrict__ xt,
                               const float* __restrict__ ve,
                               _Float16* __restrict__ X,
                               int Ns, int Nt, int Mpad) {
  size_t idx = (size_t)blockIdx.x * blockDim.x + threadIdx.x;
  size_t total = (size_t)Mpad * F_DIM;
  if (idx >= total) return;
  int row = (int)(idx / F_DIM);
  int col = (int)(idx - (size_t)row * F_DIM);
  float v;
  if (row < Ns)              v = xs[(size_t)row * F_DIM + col];
  else if (row < Ns + Nt)    v = xt[(size_t)(row - Ns) * F_DIM + col];
  else if (row == Ns + Nt)   v = ve[col];
  else                       v = 0.0f;
  X[idx] = (_Float16)v;
}

// ---------------------------------------------------------------------------
// WMMA GEMM + bias + ReLU:  D = relu(A(M x K) * W(N x K)^T + bias)
//
// Block = 256 threads = 8 waves; block owns a 128-row x 64-col output panel
// (wave w -> m-tile mblock*8+w; all waves full, EXEC all-ones).
// Per KBLK=64 step the block async-copies the 64x64 weight slice to LDS with
// global_load_async_to_lds_b128 (double-buffered, ASYNCcnt), then each wave
// runs 2 K-substeps x 4 N-tiles of v_wmma_f32_16x16x32_f16. All 8 B-fragment
// ds_load_b128s of a substep are hoisted ahead of the WMMA chain so the
// scheduler can overlap LDS latency with XDL execution (partial dscnt waits).
// ---------------------------------------------------------------------------
__global__ __launch_bounds__(256) void wmma_gemm_relu(
    const _Float16* __restrict__ A, const _Float16* __restrict__ W,
    const float* __restrict__ bias, _Float16* __restrict__ outH,
    float* __restrict__ outF, int nMblocks, int K, int Nn) {
  __shared__ __align__(16) _Float16 Bs[2][64 * BROW];

  int tid  = threadIdx.x;
  int lane = tid & 31;
  int wv   = tid >> 5;                       // 0..7
  int mblock = blockIdx.x % nMblocks;
  int ng     = blockIdx.x / nMblocks;
  int nbase  = ng * 64;
  int mtile  = mblock * 8 + wv;

  int lm = lane & 15;                        // 0..15
  int lh = lane >> 4;                        // 0..1

  // A fragment base: row M = lm, K runs {lh*8..+8} and {16+lh*8..+8}
  const _Float16* arow = A + (size_t)(mtile * 16 + lm) * K + lh * 8;

  // async staging: thread copies 32B of the 64 x KBLK weight slice
  int srow = tid >> 2;                       // 0..63 (weight row in slice)
  int sseg = tid & 3;                        // 0..3  (32B segment)
  const _Float16* wsrc = W + (size_t)(nbase + srow) * K + sseg * 16;
  unsigned ldsb[2];
  ldsb[0] = (unsigned)(uintptr_t)&Bs[0][srow * BROW + sseg * 16];
  ldsb[1] = (unsigned)(uintptr_t)&Bs[1][srow * BROW + sseg * 16];

  union Frag { v16h v; v8h h[2]; };
  v8f acc[4] = {};

  // prefill buffer 0 with K-block 0
  asm volatile("global_load_async_to_lds_b128 %0, %1, off\n\t"
               "global_load_async_to_lds_b128 %0, %1, off offset:16"
               :: "v"(ldsb[0]), "v"(wsrc) : "memory");

  int nk = K / KBLK;
  for (int ki = 0; ki < nk; ++ki) {
    int cur = ki & 1, nxt = cur ^ 1;
    int k = ki * KBLK;

    // my async writes for buffer `cur` are complete
    asm volatile("s_wait_asynccnt 0x0" ::: "memory");
    // publish buffer `cur`; also retires last round's LDS reads of `nxt`
    __syncthreads();

    // kick off next K-block into the other buffer (dummy refill on last)
    {
      int kn = (ki + 1 < nk) ? (k + KBLK) : 0;
      const _Float16* src = wsrc + kn;
      asm volatile("global_load_async_to_lds_b128 %0, %1, off\n\t"
                   "global_load_async_to_lds_b128 %0, %1, off offset:16"
                   :: "v"(ldsb[nxt]), "v"(src) : "memory");
    }

    const _Float16* bbase = &Bs[cur][0];
#pragma unroll
    for (int ss = 0; ss < 2; ++ss) {         // two 32-wide K substeps
      __builtin_prefetch(arow + k + 128, 0, 3);     // -> global_prefetch_b8
      Frag af;
      af.h[0] = *reinterpret_cast<const v8h*>(arow + k + ss * 32);
      af.h[1] = *reinterpret_cast<const v8h*>(arow + k + ss * 32 + 16);
      // hoist all B-fragment LDS loads ahead of the WMMA chain
      Frag bf[4];
#pragma unroll
      for (int t = 0; t < 4; ++t) {
        const _Float16* bp = bbase + (t * 16 + lm) * BROW + ss * 32 + lh * 8;
        bf[t].h[0] = *reinterpret_cast<const v8h*>(bp);      // ds_load_b128
        bf[t].h[1] = *reinterpret_cast<const v8h*>(bp + 16); // ds_load_b128
      }
#pragma unroll
      for (int t = 0; t < 4; ++t) {
        acc[t] = __builtin_amdgcn_wmma_f32_16x16x32_f16(
            /*neg_a=*/false, af.v, /*neg_b=*/false, bf[t].v,
            /*c_mod=*/(short)0, acc[t], /*reuse_a=*/false, /*reuse_b=*/false);
      }
    }
  }

  // D layout: lane col N = lm (+16t), VGPR r row M = r + 8*lh
#pragma unroll
  for (int t = 0; t < 4; ++t) {
    int nn = nbase + t * 16 + lm;
    float bz = bias[nn];
#pragma unroll
    for (int r = 0; r < 8; ++r) {
      int mrow = mtile * 16 + r + 8 * lh;
      float val = acc[t][r] + bz;
      val = val > 0.0f ? val : 0.0f;
      size_t o = (size_t)mrow * Nn + nn;
      if (outH) outH[o] = (_Float16)val;
      if (outF) outF[o] = val;
    }
  }
}

// ---------------------------------------------------------------------------
// Per-pair padded cdist + normalization. One 256-thread block per pair.
// Writes costs (d_out + workspace copy) and zeroes the align region.
// ---------------------------------------------------------------------------
__global__ __launch_bounds__(256) void costs_kernel(
    const float* __restrict__ Emb, const int* __restrict__ len_s,
    const int* __restrict__ len_t, int Ns, int Nt,
    float* __restrict__ out_aligns, float* __restrict__ out_costs,
    float* __restrict__ ws_costs) {
  int p = blockIdx.x, tid = threadIdx.x;
  __shared__ float sred[256];
  __shared__ float sscale;

  int n = len_s[p], m = len_t[p];
  int offs = 0, offt = 0;
  for (int q = 0; q < p; ++q) { offs += len_s[q]; offt += len_t[q]; }
  int vrow = Ns + Nt;

  float dloc[5];
  int   vloc[5];
  float lsum = 0.0f;
  int nc = 0;
  for (int cell = tid; cell < CELLS; cell += 256, ++nc) {
    int i = cell / L1, j = cell - (cell / L1) * L1;
    int ra = (i < n) ? (offs + i)      : vrow;
    int rb = (j < m) ? (Ns + offt + j) : vrow;
    const float* pa = Emb + (size_t)ra * E_DIM;
    const float* pb = Emb + (size_t)rb * E_DIM;
    float sq = 0.0f;
    for (int d = 0; d < E_DIM; ++d) { float df = pa[d] - pb[d]; sq += df * df; }
    float dd = (sq > 0.0f) ? sqrtf(sq) : 0.0f;
    int valid = (i <= n) && (j <= m);
    dloc[nc] = dd; vloc[nc] = valid;
    if (valid) lsum += dd;
  }
  sred[tid] = lsum;
  __syncthreads();
  for (int s = 128; s > 0; s >>= 1) {
    if (tid < s) sred[tid] += sred[tid + s];
    __syncthreads();
  }
  if (tid == 0) sscale = (float)(n * m) / sred[0];
  __syncthreads();
  float scale = sscale;

  nc = 0;
  for (int cell = tid; cell < CELLS; cell += 256, ++nc) {
    float c = vloc[nc] ? dloc[nc] * scale : 0.0f;
    out_costs[(size_t)p * CELLS + cell] = c;
    ws_costs [(size_t)p * CELLS + cell] = c;
    out_aligns[(size_t)p * CELLS + cell] = 0.0f;
  }
}

// ---------------------------------------------------------------------------
// Jonker-Volgenant LAP per pair (N = n+m <= 64). One 128-thread block per
// pair; C and JV state in LDS; inner column scan / argmin / dual updates are
// parallel across threads. Also emits alignment matrix and GED.
// ---------------------------------------------------------------------------
__global__ __launch_bounds__(128) void lap_kernel(
    const float* __restrict__ wcost, const int* __restrict__ len_s,
    const int* __restrict__ len_t, float* __restrict__ out_aligns,
    const float* __restrict__ out_costs, float* __restrict__ out_geds) {
  int p = blockIdx.x, tid = threadIdx.x;
  __shared__ float C[64 * 64];
  __shared__ float su[65], sv[65], sminv[65];
  __shared__ int   sway[65], spc[65], sused[65];
  __shared__ float rv[64];
  __shared__ int   ri[64];
  __shared__ float red2[128];
  __shared__ int   sj0, sj1, sdone;
  __shared__ float sdelta;

  int n = len_s[p], m = len_t[p];
  int NN = n + m;
  const float* ec = wcost + (size_t)p * CELLS;   // (n+1)x(m+1) live, stride 33

  for (int cell = tid; cell < NN * NN; cell += blockDim.x) {
    int r = cell / NN, c = cell - r * NN;
    float val;
    if (r < n) val = (c < m) ? ec[r * L1 + c]
                             : ((c - m == r) ? ec[r * L1 + m] : INF_COST);
    else       val = (c < m) ? ((r - n == c) ? ec[n * L1 + c] : INF_COST)
                             : 0.0f;
    C[r * 64 + c] = val;
  }
  for (int j = tid; j <= NN; j += blockDim.x) {
    su[j] = 0.0f; sv[j] = 0.0f; spc[j] = 0; sway[j] = 0;
  }
  __syncthreads();

  for (int i = 1; i <= NN; ++i) {
    if (tid == 0) { spc[0] = i; sj0 = 0; sdone = 0; }
    for (int j = tid; j <= NN; j += blockDim.x) { sminv[j] = BIG; sused[j] = 0; }
    __syncthreads();
    while (true) {
      int j0 = sj0;
      if (tid == 0) sused[j0] = 1;
      __syncthreads();
      int i0 = spc[j0];
      if (tid < NN) {
        int j = tid + 1;
        if (!sused[j]) {
          float cur = C[(i0 - 1) * 64 + (j - 1)] - su[i0] - sv[j];
          if (cur < sminv[j]) { sminv[j] = cur; sway[j] = j0; }
        }
      }
      __syncthreads();
      float val = BIG; int idx = 1;
      if (tid < NN) { int j = tid + 1; if (!sused[j]) { val = sminv[j]; idx = j; } }
      if (tid < 64) { rv[tid] = val; ri[tid] = idx; }
      __syncthreads();
      for (int s = 32; s > 0; s >>= 1) {
        if (tid < s && rv[tid + s] < rv[tid]) { rv[tid] = rv[tid + s]; ri[tid] = ri[tid + s]; }
        __syncthreads();
      }
      if (tid == 0) { sj1 = ri[0]; sdelta = rv[0]; }
      __syncthreads();
      float delta = sdelta;
      int j1 = sj1;
      for (int jj = tid; jj <= NN; jj += blockDim.x) {
        if (sused[jj]) { su[spc[jj]] += delta; sv[jj] -= delta; }
        else if (jj >= 1) sminv[jj] -= delta;
      }
      __syncthreads();
      if (tid == 0) { sj0 = j1; sdone = (spc[j1] == 0); }
      __syncthreads();
      if (sdone) break;
    }
    if (tid == 0) {
      int j0 = sj0;
      while (j0) { int jp = sway[j0]; spc[j0] = spc[jp]; j0 = jp; }
    }
    __syncthreads();
  }

  if (tid < NN) {
    int j = tid + 1;
    int r = spc[j] - 1;
    int c = j - 1;
    int rr = r < n ? r : n;
    int cc = c < m ? c : m;
    out_aligns[(size_t)p * CELLS + rr * L1 + cc] = 1.0f;
  }
  __syncthreads();
  if (tid == 0) out_aligns[(size_t)p * CELLS + n * L1 + m] = 0.0f;
  __syncthreads();

  float acc = 0.0f;
  for (int cell = tid; cell < CELLS; cell += blockDim.x)
    acc += out_aligns[(size_t)p * CELLS + cell] * out_costs[(size_t)p * CELLS + cell];
  red2[tid] = acc;
  __syncthreads();
  for (int s = 64; s > 0; s >>= 1) {
    if (tid < s) red2[tid] += red2[tid + s];
    __syncthreads();
  }
  if (tid == 0) out_geds[p] = red2[0] / (float)(n + m);
}

// ---------------------------------------------------------------------------
// Host launch
// ---------------------------------------------------------------------------
extern "C" void kernel_launch(void* const* d_in, const int* in_sizes, int n_in,
                              void* d_out, int out_size, void* d_ws, size_t ws_size,
                              hipStream_t stream) {
  (void)n_in; (void)out_size; (void)ws_size;
  const float* xs = (const float*)d_in[0];
  const float* xt = (const float*)d_in[1];
  const float* W1 = (const float*)d_in[2];
  const float* b1 = (const float*)d_in[3];
  const float* W2 = (const float*)d_in[4];
  const float* b2 = (const float*)d_in[5];
  const float* ve = (const float*)d_in[6];
  const int* len_s = (const int*)d_in[7];
  const int* len_t = (const int*)d_in[8];

  int Ns = in_sizes[0] / F_DIM;
  int Nt = in_sizes[1] / F_DIM;
  int Nrows = Ns + Nt + 1;
  int Mpad = (Nrows + 127) & ~127;   // 128-row panels: every wave fully active

  char* base = (char*)d_ws;
  size_t off = 0;
  auto carve = [&](size_t bytes) -> char* {
    off = (off + 255) & ~(size_t)255;
    char* ptr = base + off;
    off += bytes;
    return ptr;
  };
  _Float16* Xh    = (_Float16*)carve((size_t)Mpad * F_DIM * sizeof(_Float16));
  _Float16* W1h   = (_Float16*)carve((size_t)E_DIM * F_DIM * sizeof(_Float16));
  _Float16* W2h   = (_Float16*)carve((size_t)E_DIM * E_DIM * sizeof(_Float16));
  _Float16* Hh    = (_Float16*)carve((size_t)Mpad * E_DIM * sizeof(_Float16));
  float*    Emb   = (float*)   carve((size_t)Mpad * E_DIM * sizeof(float));
  float*    wcost = (float*)   carve((size_t)P_PAIRS * CELLS * sizeof(float));

  float* out_aligns = (float*)d_out;
  float* out_costs  = out_aligns + (size_t)P_PAIRS * CELLS;
  float* out_geds   = out_costs  + (size_t)P_PAIRS * CELLS;

  {
    int nel = E_DIM * F_DIM;
    f32_to_f16_kernel<<<(nel + 255) / 256, 256, 0, stream>>>(W1, W1h, nel);
  }
  {
    int nel = E_DIM * E_DIM;
    f32_to_f16_kernel<<<(nel + 255) / 256, 256, 0, stream>>>(W2, W2h, nel);
  }
  {
    size_t total = (size_t)Mpad * F_DIM;
    int blocks = (int)((total + 255) / 256);
    build_x_kernel<<<blocks, 256, 0, stream>>>(xs, xt, ve, Xh, Ns, Nt, Mpad);
  }
  {
    int nMblocks = Mpad / 128;
    int blocks = nMblocks * (E_DIM / 64);
    wmma_gemm_relu<<<blocks, 256, 0, stream>>>(Xh, W1h, b1, Hh, nullptr,
                                               nMblocks, F_DIM, E_DIM);
    wmma_gemm_relu<<<blocks, 256, 0, stream>>>(Hh, W2h, b2, nullptr, Emb,
                                               nMblocks, E_DIM, E_DIM);
  }
  costs_kernel<<<P_PAIRS, 256, 0, stream>>>(Emb, len_s, len_t, Ns, Nt,
                                            out_aligns, out_costs, wcost);
  lap_kernel<<<P_PAIRS, 128, 0, stream>>>(wcost, len_s, len_t,
                                          out_aligns, out_costs, out_geds);
}
